// CNN_65867618451728
// MI455X (gfx1250) — matrix operand order repaired and checked by reading, not compiled, not run on hardware
//
#include <hip/hip_runtime.h>
#include <hip/hip_bf16.h>
#include <math.h>
#include <stdint.h>

// ---------------------------------------------------------------------------
// SE(3)-equivariant 3D CNN for MI455X (gfx1250, wave32, WMMA).
// Convs are implicit GEMMs on v_wmma_f32_16x16x32_f16 (f16 in, f32 acc).
// - activations stored with built-in zero halo (pad 3) -> branchless A gather
// - per-element incremental (ci,tap) K decomposition + LDS spatial-offset LUT
// - B filter tile staged through LDS with async global->LDS DMA (ASYNCcnt),
//   double buffered so the copy overlaps the WMMA stream
// All intermediates fit in the 192MB L2 -> compute bound -> feed the WMMA pipe.
// ---------------------------------------------------------------------------

typedef _Float16 half_t;
typedef _Float16 v16h __attribute__((ext_vector_type(16)));
typedef float    v8f  __attribute__((ext_vector_type(8)));

union U32H2 { uint32_t u; _Float16 h[2]; };

#define EPSV 1e-5f

// ----------------------------- basis helpers ------------------------------
__device__ __forceinline__ void basis_at(int t, float hat[4], float xh[3], float& mask) {
    int kd = t / 25; int r = t - kd * 25;
    int kh = r / 5;  int kw = r - kh * 5;
    float ox = (float)kd - 2.f, oy = (float)kh - 2.f, oz = (float)kw - 2.f;
    float rr = sqrtf(ox * ox + oy * oy + oz * oz);
    #pragma unroll
    for (int k = 0; k < 4; ++k) hat[k] = fmaxf(1.f - fabsf(rr - (float)k), 0.f);
    float inv = (rr > 0.f) ? (1.f / rr) : 0.f;
    xh[0] = ox * inv; xh[1] = oy * inv; xh[2] = oz * inv;
    mask = (rr > 0.f) ? 1.f : 0.f;
}

__device__ __forceinline__ float ang1_val(int i, int j, const float* xh) {
    if (i == j) return 0.f;
    if (i == 0 && j == 1) return  xh[2];
    if (i == 1 && j == 0) return -xh[2];
    if (i == 1 && j == 2) return  xh[0];
    if (i == 2 && j == 1) return -xh[0];
    if (i == 2 && j == 0) return  xh[1];
    return -xh[1]; // i==0, j==2
}

// K1[(oc*3+j)][ci*125+t] ; rows padded to Kpad halves (16B) for async b128
__global__ void build_filt1(const float* __restrict__ w1, half_t* __restrict__ f,
                            int s_in, int Ktot, int Kpad) {
    int row = blockIdx.x;            // 0..vec*3-1
    int oc = row / 3, j = row - oc * 3;
    for (int k = threadIdx.x; k < Kpad; k += blockDim.x) {
        float out = 0.f;
        if (k < Ktot) {
            int ci = k / 125; int t = k - ci * 125;
            float hat[4], xh[3], mask;
            basis_at(t, hat, xh, mask);
            float R = 0.f;
            #pragma unroll
            for (int q = 0; q < 4; ++q) R += w1[(oc * s_in + ci) * 4 + q] * hat[q];
            out = xh[j] * R;
        }
        f[(size_t)row * Kpad + k] = (half_t)out;
    }
}

// K2[so][c2*125+t]: c2 < vec*3 -> KA (xhat), else KB (angular tensors)
__global__ void build_filt2(const float* __restrict__ wA, const float* __restrict__ wB,
                            half_t* __restrict__ f, int vec, int Ktot, int Kpad) {
    int so = blockIdx.x;
    for (int k = threadIdx.x; k < Kpad; k += blockDim.x) {
        float val = 0.f;
        if (k < Ktot) {
            int c2 = k / 125; int t = k - c2 * 125;
            float hat[4], xh[3], mask;
            basis_at(t, hat, xh, mask);
            if (c2 < vec * 3) {
                int oc = c2 / 3, j = c2 - oc * 3;
                float R = 0.f;
                #pragma unroll
                for (int q = 0; q < 4; ++q) R += wA[(so * vec + oc) * 4 + q] * hat[q];
                val = R * xh[j];
            } else {
                int c9 = c2 - vec * 3;
                int oc = c9 / 9, ij = c9 - oc * 9;
                int i = ij / 3, jj = ij - i * 3;
                float R0 = 0.f, R1 = 0.f, R2 = 0.f;
                #pragma unroll
                for (int q = 0; q < 4; ++q) {
                    R0 += wB[((so * vec + oc) * 3 + 0) * 4 + q] * hat[q];
                    R1 += wB[((so * vec + oc) * 3 + 1) * 4 + q] * hat[q];
                    R2 += wB[((so * vec + oc) * 3 + 2) * 4 + q] * hat[q];
                }
                float a0 = (i == jj) ? 1.f : 0.f;
                float a1 = ang1_val(i, jj, xh);
                float a2 = xh[i] * xh[jj] - ((i == jj) ? (mask / 3.f) : 0.f);
                val = R0 * a0 + R1 * a1 + R2 * a2;
            }
        }
        f[(size_t)so * Kpad + k] = (half_t)val;
    }
}

// --------------------------- implicit GEMM conv ----------------------------
// in: zero-halo padded f16 [B][Cin][Dp^3]; filt: f16 [Cout][Kpad]
// out[M,N] = im2col(in)[M,K] * filt^T[K,N], M = B*Dout^3, N = Cout, K = Cin*125
template <int STRIDE, bool OUTF32>
__global__ __launch_bounds__(256) void conv_wmma(
        const half_t* __restrict__ in, const half_t* __restrict__ filt,
        void* __restrict__ outp, int B, int Cin, int Dp, int Dout,
        int Cout, int Ktot, int Kpad) {
    __shared__ int lutS[128];
    __shared__ __align__(16) half_t bstage[2][16 * 128];   // double-buffered B tile

    const int tid = threadIdx.x;
    if (tid < 125) {  // spatial-offset LUT: tap -> element offset in padded volume
        int kd = tid / 25, r = tid - kd * 25;
        int kh = r / 5, kw = r - kh * 5;
        lutS[tid] = (kd * Dp + kh) * Dp + kw;
    }

    const int lane = tid & 31;
    const int wv   = tid >> 5;
    const int g    = lane >> 4;
    const int l16  = lane & 15;
    const int Dout2 = Dout * Dout, Dout3 = Dout2 * Dout;
    const int Dp2 = Dp * Dp, Dp3 = Dp2 * Dp;
    const int M = B * Dout3;

    const int Mbase = (blockIdx.x * 8 + wv) * 16;
    const int Nbase = blockIdx.y * 16;

    // Per-lane A row (clamped; no early exit so barriers stay uniform).
    int mrow = Mbase + l16; if (mrow >= M) mrow = M - 1;
    int b  = mrow / Dout3; int r = mrow - b * Dout3;
    int od = r / Dout2;    r -= od * Dout2;
    int oh = r / Dout;     int ow = r - oh * Dout;
    const size_t inBase = (size_t)b * Cin * Dp3 +
        ((size_t)(od * STRIDE) * Dp + (size_t)(oh * STRIDE)) * Dp + (size_t)(ow * STRIDE);

    // Incremental per-element K decomposition: k = ci*125 + tap, k += 32/step.
    int tap[16]; int ciOff[16];
    #pragma unroll
    for (int e = 0; e < 16; ++e) {
        int vv = e >> 1, h = e & 1;
        tap[e]   = ((vv < 4) ? (2 * vv) : (8 + 2 * vv)) + 8 * g + h;  // ISA A layout
        ciOff[e] = 0;
    }

    // async DMA one 128-K chunk of the 16-row B tile into LDS (b128 per thread)
    auto stage = [&](int buf, int k0) {
        int rrow = tid >> 4;
        int nrow = Nbase + rrow; if (nrow >= Cout) nrow = Cout - 1;
        const half_t* src = filt + (size_t)nrow * Kpad + k0 + (tid & 15) * 8;
        uint32_t ldsoff = (uint32_t)(uintptr_t)&bstage[buf][rrow * 128 + (tid & 15) * 8];
        asm volatile("global_load_async_to_lds_b128 %0, %1, off"
                     :: "v"(ldsoff), "v"(src) : "memory");
    };

    __syncthreads();        // LUT visible
    stage(0, 0);

    v8f acc = {};
    const int Nc = (Ktot + 127) >> 7;
    for (int c = 0; c < Nc; ++c) {
        asm volatile("s_wait_asynccnt 0x0" ::: "memory");   // my chunk landed
        __syncthreads();                                    // everyone's chunk landed
        if (c + 1 < Nc) stage((c + 1) & 1, (c + 1) << 7);   // overlap next DMA
        const half_t* bs = &bstage[c & 1][0];
        #pragma unroll
        for (int s = 0; s < 4; ++s) {
            const int k0 = (c << 7) + s * 32;
            if (k0 >= Ktot) break;                          // uniform
            // B fragment from LDS: K rows striped 2-per-VGPR, ISA table order (+8 rows if lane>=16, second 16 K in VGPRs 4-7)
            v16h bm;
            #pragma unroll
            for (int vv = 0; vv < 8; ++vv) {
                int kl = s * 32 + ((vv < 4) ? (2 * vv) : (8 + 2 * vv)) + 8 * g;
                U32H2 p; p.u = *(const uint32_t*)(bs + l16 * 128 + kl);
                bm[2 * vv] = p.h[0]; bm[2 * vv + 1] = p.h[1];
            }
            v16h a;
            if (k0 + 32 <= Ktot) {                          // hot path: no guards
                #pragma unroll
                for (int e = 0; e < 16; ++e) {
                    a[e] = in[inBase + (size_t)ciOff[e] + (size_t)lutS[tap[e]]];
                    int t2 = tap[e] + 32;
                    int w = (t2 >= 125);
                    tap[e]   = w ? t2 - 125 : t2;
                    ciOff[e] += w ? Dp3 : 0;
                }
            } else {                                        // K tail
                #pragma unroll
                for (int e = 0; e < 16; ++e) {
                    int vv = e >> 1, h = e & 1;
                    int koff = ((vv < 4) ? (2 * vv) : (8 + 2 * vv)) + 8 * g + h;
                    half_t val = (half_t)0;
                    if (k0 + koff < Ktot)
                        val = in[inBase + (size_t)ciOff[e] + (size_t)lutS[tap[e]]];
                    a[e] = val;
                    int t2 = tap[e] + 32;
                    int w = (t2 >= 125);
                    tap[e]   = w ? t2 - 125 : t2;
                    ciOff[e] += w ? Dp3 : 0;
                }
            }
            acc = __builtin_amdgcn_wmma_f32_16x16x32_f16(
                false, a, false, bm, (short)0, acc, false, false);
        }
    }

    // D store: VGPR j <-> row (j + 8*g), col = lane&15 (unpadded output layout)
    #pragma unroll
    for (int j = 0; j < 8; ++j) {
        int m = Mbase + j + 8 * g;
        int n = Nbase + l16;
        if (m < M && n < Cout) {
            int b2 = m / Dout3; int r2 = m - b2 * Dout3;
            int od2 = r2 / Dout2; r2 -= od2 * Dout2;
            int oh2 = r2 / Dout;  int ow2 = r2 - oh2 * Dout;
            size_t idx = ((size_t)b2 * Cout + n) * Dout3 +
                         (size_t)od2 * Dout2 + oh2 * Dout + ow2;
            if (OUTF32) ((float*)outp)[idx] = acc[j];
            else        ((half_t*)outp)[idx] = (half_t)acc[j];
        }
    }
}

// ----------------------- elementwise tensor product ------------------------
// v: unpadded f16 [B][vec*3][Dm^3] -> vt: padded f16 [B][vec*12][Dpm^3]
__global__ void tp_kernel(const half_t* __restrict__ v, half_t* __restrict__ vt,
                          int vec, int Dm, int Dpm, int total) {
    int idx = blockIdx.x * blockDim.x + threadIdx.x;
    if (idx >= total) return;
    const int Dm2 = Dm * Dm, Dm3 = Dm2 * Dm;
    const int Dpm2 = Dpm * Dpm, Dpm3 = Dpm2 * Dpm;
    int b = idx / Dm3, p = idx - b * Dm3;
    int z = p / Dm2; int r = p - z * Dm2;
    int y = r / Dm;  int x = r - y * Dm;
    int pout = ((z + 3) * Dpm + (y + 3)) * Dpm + (x + 3);
    const int C1 = vec * 3, C2 = vec * 12;
    const size_t ib = (size_t)b * C1 * Dm3 + p;
    const size_t ob = (size_t)b * C2 * Dpm3 + pout;
    for (int c = 0; c < vec; ++c) {
        float xv[3];
        #pragma unroll
        for (int i = 0; i < 3; ++i) {
            xv[i] = (float)v[ib + (size_t)(c * 3 + i) * Dm3];
            vt[ob + (size_t)(c * 3 + i) * Dpm3] = (half_t)xv[i];
        }
        #pragma unroll
        for (int i = 0; i < 3; ++i)
            #pragma unroll
            for (int j = 0; j < 3; ++j)
                vt[ob + (size_t)(C1 + c * 9 + i * 3 + j) * Dpm3] = (half_t)(xv[i] * xv[j]);
    }
}

// ------------------------------ batch norm --------------------------------
__global__ void bn_stats(const float* __restrict__ x, float* __restrict__ st,
                         int B, int C, int sp3) {
    const int c = blockIdx.x;
    const int n = B * sp3;
    float s = 0.f, q = 0.f;
    for (int i = threadIdx.x; i < n; i += blockDim.x) {
        int b = i / sp3, p = i - b * sp3;
        float val = x[((size_t)b * C + c) * sp3 + p];
        s += val; q += val * val;
    }
    __shared__ float rs[1024], rq[1024];
    rs[threadIdx.x] = s; rq[threadIdx.x] = q;
    __syncthreads();
    for (int off = blockDim.x >> 1; off > 0; off >>= 1) {
        if (threadIdx.x < off) {
            rs[threadIdx.x] += rs[threadIdx.x + off];
            rq[threadIdx.x] += rq[threadIdx.x + off];
        }
        __syncthreads();
    }
    if (threadIdx.x == 0) {
        float mean = rs[0] / (float)n;
        st[c * 2 + 0] = mean;
        st[c * 2 + 1] = rq[0] / (float)n - mean * mean;
    }
}

// x: unpadded f32 [B][C][Do^3] -> out: padded f16 [B][C][Dpo^3] interior
__global__ void bn_apply(const float* __restrict__ x, const float* __restrict__ st,
                         const float* __restrict__ gamma, const float* __restrict__ beta,
                         const float* __restrict__ bias, int relu,
                         half_t* __restrict__ out, int total, int C, int Do, int Dpo) {
    int i = blockIdx.x * blockDim.x + threadIdx.x;
    if (i >= total) return;
    const int Do2 = Do * Do, Do3 = Do2 * Do;
    const int Dpo2 = Dpo * Dpo, Dpo3 = Dpo2 * Dpo;
    int bc = i / Do3, p = i - bc * Do3;
    int c = bc % C;
    int z = p / Do2; int r = p - z * Do2;
    int y = r / Do;  int xx = r - y * Do;
    float m = st[c * 2 + 0], v = st[c * 2 + 1];
    float val = (x[i] - m) * rsqrtf(v + EPSV) * gamma[c] + beta[c];
    if (relu) val = fmaxf(val + bias[c], 0.f);
    out[(size_t)bc * Dpo3 + ((size_t)(z + 3) * Dpo + (y + 3)) * Dpo + (xx + 3)] = (half_t)val;
}

// input BN (C==1, no affine): normalize f32 -> padded f16
__global__ void in_apply(const float* __restrict__ x, const float* __restrict__ st,
                         half_t* __restrict__ out, int D, int Dp, int total) {
    int i = blockIdx.x * blockDim.x + threadIdx.x;
    if (i >= total) return;
    const int D2 = D * D, D3 = D2 * D;
    const int Dp2 = Dp * Dp, Dp3 = Dp2 * Dp;
    int b = i / D3, p = i - b * D3;
    int z = p / D2; int r = p - z * D2;
    int y = r / D;  int xx = r - y * D;
    float val = (x[i] - st[0]) * rsqrtf(st[1] + EPSV);
    out[(size_t)b * Dp3 + ((size_t)(z + 3) * Dp + (y + 3)) * Dp + (xx + 3)] = (half_t)val;
}

// ------------------------------- pool + head -------------------------------
// s: padded f16 [B][10][Dp^3], average interior Do^3
__global__ void pool_kernel(const half_t* __restrict__ s, float* __restrict__ pooled,
                            int Do, int Dp) {
    int bc = blockIdx.x;  // b*10 + c
    const int Do2 = Do * Do, Do3 = Do2 * Do;
    const int Dp2 = Dp * Dp, Dp3 = Dp2 * Dp;
    __shared__ float red[64];
    float sum = 0.f;
    for (int p = threadIdx.x; p < Do3; p += 64) {
        int z = p / Do2; int r = p - z * Do2;
        int y = r / Do;  int x = r - y * Do;
        sum += (float)s[(size_t)bc * Dp3 + ((size_t)(z + 3) * Dp + (y + 3)) * Dp + (x + 3)];
    }
    red[threadIdx.x] = sum;
    __syncthreads();
    for (int off = 32; off > 0; off >>= 1) {
        if (threadIdx.x < off) red[threadIdx.x] += red[threadIdx.x + off];
        __syncthreads();
    }
    if (threadIdx.x == 0) pooled[bc] = red[0] / (float)Do3;
}

__global__ void head_kernel(const float* __restrict__ pooled,
                            const float* __restrict__ gamma, const float* __restrict__ beta,
                            float* __restrict__ out) {
    __shared__ float m[10], vv[10];
    int tid = threadIdx.x;
    if (tid < 10) {
        float s = 0.f, q = 0.f;
        for (int b = 0; b < 32; ++b) {
            float p = pooled[b * 10 + tid];
            s += p; q += p * p;
        }
        float mean = s / 32.f;
        m[tid] = mean;
        vv[tid] = q / 32.f - mean * mean;
    }
    __syncthreads();
    for (int i = tid; i < 320; i += blockDim.x) {
        int c = i % 10;
        out[i] = gamma[c] * (pooled[i] - m[c]) * rsqrtf(vv[c] + EPSV) + beta[c];
    }
}

// --------------------------------- driver ----------------------------------
extern "C" void kernel_launch(void* const* d_in, const int* in_sizes, int n_in,
                              void* d_out, int out_size, void* d_ws, size_t ws_size,
                              hipStream_t stream) {
    (void)in_sizes; (void)n_in; (void)out_size; (void)ws_size;

    const float* x = (const float*)d_in[0];
    const float *w1[5], *wA[5], *wB[5], *bias[5], *gamma[5], *beta[5];
    for (int i = 0; i < 5; ++i) {
        w1[i]    = (const float*)d_in[1 + 6 * i];
        wA[i]    = (const float*)d_in[2 + 6 * i];
        wB[i]    = (const float*)d_in[3 + 6 * i];
        bias[i]  = (const float*)d_in[4 + 6 * i];
        gamma[i] = (const float*)d_in[5 + 6 * i];
        beta[i]  = (const float*)d_in[6 + 6 * i];
    }
    const float* gamma_out = (const float*)d_in[31];
    const float* beta_out  = (const float*)d_in[32];

    // Workspace layout (~418 MB); hot working set stays L2-resident on MI455X.
    char* ws = (char*)d_ws;
    float*  stCh   = (float*)(ws + 64);                 // per-channel BN stats
    float*  pooled = (float*)(ws + 512);                // [32,10]
    half_t* F1 = (half_t*)(ws + 4096);                  // K1, rows padded (cap 256KB)
    half_t* F2 = (half_t*)(ws + 4096 + (256 << 10));    // K2, rows padded (cap ~1.75MB)
    size_t off = (size_t)2 << 20;
    half_t* S0 = (half_t*)(ws + off); off += (size_t)48  << 20;
    half_t* S1 = (half_t*)(ws + off); off += (size_t)48  << 20;
    half_t* V  = (half_t*)(ws + off); off += (size_t)32  << 20;
    half_t* VT = (half_t*)(ws + off); off += (size_t)192 << 20;
    float*  C2 = (float*)(ws + off);                    // 96 MB f32 conv2 out

    static const int s_in[5]  = {1, 8, 16, 16, 16};
    static const int s_out[5] = {8, 16, 16, 16, 10};
    static const int vecs[5]  = {4, 12, 16, 16, 13};
    static const int Din_[5]  = {64, 35, 21, 14, 10};  // conv1 input spatial (unpadded)
    static const int Dmid[5]  = {33, 19, 12, 8, 6};    // conv1 out / conv2 in
    static const int Dout_[5] = {35, 21, 14, 10, 8};   // conv2 out spatial

    // ---- input batch norm (single channel, no affine) -> padded f16 ----
    const int n0 = 32 * 64 * 64 * 64;
    const int Dp0 = 64 + 6;
    bn_stats<<<1, 1024, 0, stream>>>(x, stCh, 32, 1, 64 * 64 * 64);
    hipMemsetAsync(S0, 0, (size_t)32 * Dp0 * Dp0 * Dp0 * sizeof(half_t), stream);
    in_apply<<<(n0 + 255) / 256, 256, 0, stream>>>(x, stCh, S0, 64, Dp0, n0);

    half_t* Scur = S0;
    half_t* Snxt = S1;
    for (int i = 0; i < 5; ++i) {
        const int si = s_in[i], so = s_out[i], ve = vecs[i];
        const int C1 = ve * 3, C2c = ve * 12;
        const int Di = Din_[i], Dm = Dmid[i], Do = Dout_[i];
        const int Dpi = Di + 6, Dpm = Dm + 6, Dpo = Do + 6;
        const int K1 = si * 125, K2k = C2c * 125;
        const int K1p = (K1 + 7) & ~7, K2p = (K2k + 7) & ~7;
        const int sp_m = Dm * Dm * Dm, sp_o = Do * Do * Do;

        build_filt1<<<C1, 128, 0, stream>>>(w1[i], F1, si, K1, K1p);
        build_filt2<<<so, 128, 0, stream>>>(wA[i], wB[i], F2, ve, K2k, K2p);

        // conv1: scalar -> vector, stride 2, padded f16 in, unpadded f16 out
        const int M1 = 32 * sp_m;
        dim3 g1((M1 + 127) / 128, (C1 + 15) / 16);
        conv_wmma<2, false><<<g1, 256, 0, stream>>>(Scur, F1, V, 32, si, Dpi, Dm, C1, K1, K1p);

        // tensor product -> padded vt (zero halo first)
        hipMemsetAsync(VT, 0, (size_t)32 * C2c * Dpm * Dpm * Dpm * sizeof(half_t), stream);
        tp_kernel<<<(M1 + 255) / 256, 256, 0, stream>>>(V, VT, ve, Dm, Dpm, M1);

        // conv2: (repr3 ++ repr3x3) -> scalar, stride 1, f32 out
        const int M2 = 32 * sp_o;
        dim3 g2((M2 + 127) / 128, (so + 15) / 16);
        conv_wmma<1, true><<<g2, 256, 0, stream>>>(VT, F2, C2, 32, C2c, Dpm, Do, so, K2k, K2p);

        // batch norm (+bias+relu for all but last block) -> padded f16
        bn_stats<<<so, 1024, 0, stream>>>(C2, stCh, 32, so, sp_o);
        hipMemsetAsync(Snxt, 0, (size_t)32 * so * Dpo * Dpo * Dpo * sizeof(half_t), stream);
        const int tot2 = 32 * so * sp_o;
        bn_apply<<<(tot2 + 255) / 256, 256, 0, stream>>>(
            C2, stCh, gamma[i], beta[i], bias[i], (i < 4) ? 1 : 0, Snxt, tot2, so, Do, Dpo);

        half_t* t = Scur; Scur = Snxt; Snxt = t;
    }

    // global average pool over 8^3 interior and output batch norm head
    pool_kernel<<<32 * 10, 64, 0, stream>>>(Scur, pooled, 8, 14);
    head_kernel<<<1, 64, 0, stream>>>(pooled, gamma_out, beta_out, (float*)d_out);
}
